// SelfAttention_5875515261371
// MI455X (gfx1250) — matrix operand order repaired and checked by reading, not compiled, not run on hardware
//
#include <hip/hip_runtime.h>

// ---------------------------------------------------------------------------
// Self-attention forward for MI455X (gfx1250, wave32, WMMA).
//
//   Q = x@Wq + bq ; K = x@Wk + bk ; V = x@Wv + bv        (bf16 WMMA, f32 acc)
//   scores  = Q @ K^T / sqrt(D)                          (bf16 WMMA, f32 acc)
//   weights = softmax(scores)                            (f32, in-place -> bf16)
//   out     = weights @ V                                (bf16 WMMA, f32 acc)
//
// Intermediates (~130 MB) are L2-resident (192 MB L2). GEMMs are bf16 WMMA
// with f32 accumulation; B-operand tiles are staged to LDS with CDNA5
// GLOBAL_LOAD_ASYNC_TO_LDS (ASYNCcnt-tracked, triple-buffered, one barrier
// per K-slab, rotating AS3 pointers so no per-iteration addrspace casts).
// ---------------------------------------------------------------------------

#define B_ 4
#define S_ 2048
#define D_ 1024

typedef __attribute__((ext_vector_type(16))) __bf16 v16bf;
typedef __attribute__((ext_vector_type(8)))  __bf16 v8bf;
typedef __attribute__((ext_vector_type(8)))  float  v8f;
typedef __attribute__((ext_vector_type(4)))  int    v4i;

#define AS1 __attribute__((address_space(1)))
#define AS3 __attribute__((address_space(3)))

#if defined(__gfx1250__) &&                                            \
    __has_builtin(__builtin_amdgcn_global_load_async_to_lds_b128) &&   \
    __has_builtin(__builtin_amdgcn_s_wait_asynccnt)
#define USE_ASYNC_LDS 1
#else
#define USE_ASYNC_LDS 0
#endif

__device__ __forceinline__ unsigned short f32_to_bf16(float f) {
  unsigned int u = __float_as_uint(f);
  u += 0x7fffu + ((u >> 16) & 1u);   // round-to-nearest-even
  return (unsigned short)(u >> 16);
}

// Stage one 16-byte chunk global -> LDS (async direct-to-LDS when available).
// Builtin prototype (from hipcc diagnostic): (int4 AS1*, int4 AS3*, Imm, Imm).
__device__ __forceinline__ void stage_b_chunk(AS1 const unsigned short* g,
                                              AS3 unsigned short* l) {
#if USE_ASYNC_LDS
  __builtin_amdgcn_global_load_async_to_lds_b128(
      (AS1 v4i*)(AS1 unsigned short*)g, (AS3 v4i*)l,
      /*imm offset*/ 0, /*cpol*/ 0);
#else
  *(AS3 v8bf*)l = *(const AS1 v8bf*)g;
#endif
}

template <int N>
__device__ __forceinline__ void wait_async_slab() {
#if USE_ASYNC_LDS
  __builtin_amdgcn_s_wait_asynccnt(N);
#endif
}

// -------------------------- conversion kernels -----------------------------

__global__ __launch_bounds__(256) void cvt_f32_to_bf16(
    const float* __restrict__ in, unsigned short* __restrict__ out, int n) {
  for (int i = blockIdx.x * blockDim.x + threadIdx.x; i < n;
       i += gridDim.x * blockDim.x)
    out[i] = f32_to_bf16(in[i]);
}

// Wt[n][k] = W[k][n]  (so GEMM "B" fragment loads are contiguous along K)
__global__ __launch_bounds__(256) void transpose_cvt_w(
    const float* __restrict__ W, unsigned short* __restrict__ Wt) {
  int i = blockIdx.x * blockDim.x + threadIdx.x;
  if (i < D_ * D_) {
    int n = i / D_, k = i % D_;
    Wt[i] = f32_to_bf16(W[(size_t)k * D_ + n]);
  }
}

// ------------------------------ WMMA GEMM ----------------------------------
// C[M x N] = A[M x K] * Bt[N x K]^T   (A, Bt bf16 row-major; f32 accumulate)
// OUT_MODE 0: f32  C[m*ldc + n] = acc * alpha
// OUT_MODE 1: bf16 C[m*ldc + n] = acc + bias[n]
// OUT_MODE 2: bf16 C[n*ldc + m] = acc + bias[n]   (transposed store, for V)
//
// Block = 256 threads (8 wave32). Block tile 256(M) x 64(N); each wave owns a
// 32-row strip (2 A-fragments) x four 16-col tiles => 8 WMMAs per 32-wide
// K-slab, each B fragment reused by two WMMAs. Bt slab staged in LDS,
// triple-buffered, 40-element row pitch (bank-conflict-free b128 reads).
#define BROW 40              // padded LDS row pitch in bf16 elements (80 B)
#define SLAB (64 * BROW)     // one staged slab, in bf16 elements

template <int OUT_MODE>
__global__ __launch_bounds__(256) void gemm_bf16_wmma(
    const unsigned short* __restrict__ A, int lda, size_t strideA,
    const unsigned short* __restrict__ Bt, int ldb, size_t strideB,
    void* __restrict__ C, int ldc, size_t strideC,
    const float* __restrict__ bias, int K, float alpha) {
  __shared__ __align__(128) unsigned short ldsB[3 * SLAB];  // 3 x 5 KB

  const int tid  = threadIdx.x;
  const int lane = tid & 31;
  const int wave = tid >> 5;
  const int half = lane >> 4;   // 0: lanes 0-15, 1: lanes 16-31
  const int l16  = lane & 15;
  const int b    = blockIdx.z;

  A  += (size_t)b * strideA;
  Bt += (size_t)b * strideB;

  const int m0 = blockIdx.x * 256 + wave * 32;  // wave's 32 output rows
  const int n0 = blockIdx.y * 64;               // block's 64 output cols

  v8f acc[2][4] = {};

  const unsigned short* Arow[2];
  Arow[0] = A + (size_t)(m0 + l16) * lda;
  Arow[1] = A + (size_t)(m0 + 16 + l16) * lda;

  // LDS staging: thread t copies one 16-B chunk of the 64x32 Bt slab.
  const int srow = tid >> 2;        // 0..63   (output column within tile)
  const int schk = (tid & 3) * 8;   // 0,8,16,24 (k offset within slab)

  // One addrspacecast each, hoisted; all loop address math is AS-local.
  AS3 unsigned short* lds3 = (AS3 unsigned short*)&ldsB[0];
  AS1 const unsigned short* BstG =
      (AS1 const unsigned short*)(Bt + (size_t)(n0 + srow) * ldb + schk);

  const int lofs  = srow * BROW + schk;        // this thread's staging slot
  const int rdofs = l16 * BROW + half * 16;    // this lane's B-fragment base

  // Rotating triple-buffer pointers (3 reg moves/iter instead of %3 + select).
  AS3 unsigned short* st0 = lds3 + 0 * SLAB + lofs;
  AS3 unsigned short* st1 = lds3 + 1 * SLAB + lofs;
  AS3 unsigned short* st2 = lds3 + 2 * SLAB + lofs;
  AS3 const unsigned short* rd0 = lds3 + 0 * SLAB + rdofs;
  AS3 const unsigned short* rd1 = lds3 + 1 * SLAB + rdofs;
  AS3 const unsigned short* rd2 = lds3 + 2 * SLAB + rdofs;

  // 8 WMMAs on the slab at LDS read base `rb`, A fragments at k-offset k0.
  auto compute_slab = [&](int k0, AS3 const unsigned short* rb) {
    // A fragments, contiguous-K layout (matches B):
    //   lanes 0-15: K 0-15 ; lanes 16-31: K 16-31
    union { v16bf v; v8bf h[2]; } af[2];
#pragma unroll
    for (int hh = 0; hh < 2; ++hh) {
      const unsigned short* ar = Arow[hh] + k0;
      af[hh].h[0] = *(const v8bf*)(ar + half * 16);
      af[hh].h[1] = *(const v8bf*)(ar + half * 16 + 8);
    }
    // B fragments: lane = column n; lanes 0-15 K0-15, lanes 16-31 K16-31.
    v16bf bt[4];
#pragma unroll
    for (int t = 0; t < 4; ++t)
      bt[t] = *(const AS3 v16bf*)(rb + t * 16 * BROW);
#pragma unroll
    for (int t = 0; t < 4; ++t) {
      acc[0][t] = __builtin_amdgcn_wmma_f32_16x16x32_bf16(
          false, af[0].v, false, bt[t], (short)0, acc[0][t], false, false);
      acc[1][t] = __builtin_amdgcn_wmma_f32_16x16x32_bf16(
          false, af[1].v, false, bt[t], (short)0, acc[1][t], false, false);
    }
  };

  const int nk = K >> 5;            // 32-wide K slabs
  stage_b_chunk(BstG, st0);         // preload slab 0

  for (int it = 0; it < nk - 1; ++it) {
    const int k0 = it << 5;
    stage_b_chunk(BstG + k0 + 32, st1);  // prefetch slab i+1 (async)
    wait_async_slab<1>();     // older (slab i) async loads complete in order
    __syncthreads();          // all threads' chunks of slab i visible
    __builtin_prefetch(Arow[0] + k0 + 32, 0, 3);
    compute_slab(k0, rd0);
    // rotate triple buffers
    AS3 unsigned short* ts = st0; st0 = st1; st1 = st2; st2 = ts;
    AS3 const unsigned short* tr = rd0; rd0 = rd1; rd1 = rd2; rd2 = tr;
  }
  wait_async_slab<0>();
  __syncthreads();
  compute_slab((nk - 1) << 5, rd0);

  // C/D layout: VGPR r, lanes 0-15 -> (M=r, N=lane); lanes 16-31 -> (M=8+r).
#pragma unroll
  for (int hh = 0; hh < 2; ++hh) {
#pragma unroll
    for (int t = 0; t < 4; ++t) {
      const int n = n0 + t * 16 + l16;
      const float bv = (OUT_MODE == 0) ? 0.0f : bias[n];
#pragma unroll
      for (int r = 0; r < 8; ++r) {
        const int m = m0 + hh * 16 + half * 8 + r;
        const float v = acc[hh][t][r];
        if (OUT_MODE == 0) {
          ((float*)C)[(size_t)b * strideC + (size_t)m * ldc + n] = v * alpha;
        } else if (OUT_MODE == 1) {
          ((unsigned short*)C)[(size_t)b * strideC + (size_t)m * ldc + n] =
              f32_to_bf16(v + bv);
        } else {
          ((unsigned short*)C)[(size_t)b * strideC + (size_t)n * ldc + m] =
              f32_to_bf16(v + bv);
        }
      }
    }
  }
}

// ------------------------------- softmax -----------------------------------
// One block per (b,q) row of length S_. Reads 2048 f32, writes 2048 bf16
// IN PLACE over the start of the same row (reads are barrier-ordered first).
__global__ __launch_bounds__(256) void softmax_rows(float* __restrict__ scores) {
  __shared__ float red[256];
  float* rp = scores + (size_t)blockIdx.x * S_;
  const int tid = threadIdx.x;

  float v[8];
  float mx = -3.4e38f;
#pragma unroll
  for (int i = 0; i < 8; ++i) {
    v[i] = rp[tid + i * 256];
    mx = fmaxf(mx, v[i]);
  }
  red[tid] = mx;
  __syncthreads();
  for (int s = 128; s > 0; s >>= 1) {
    if (tid < s) red[tid] = fmaxf(red[tid], red[tid + s]);
    __syncthreads();
  }
  mx = red[0];
  __syncthreads();

  float sum = 0.f;
#pragma unroll
  for (int i = 0; i < 8; ++i) { v[i] = __expf(v[i] - mx); sum += v[i]; }
  red[tid] = sum;
  __syncthreads();
  for (int s = 128; s > 0; s >>= 1) {
    if (tid < s) red[tid] += red[tid + s];
    __syncthreads();
  }
  const float inv = 1.0f / red[0];

  unsigned short* wp = (unsigned short*)rp;  // bf16 weights overlay the row
#pragma unroll
  for (int i = 0; i < 8; ++i) wp[tid + i * 256] = f32_to_bf16(v[i] * inv);
}

// ------------------------------- launcher ----------------------------------

extern "C" void kernel_launch(void* const* d_in, const int* in_sizes, int n_in,
                              void* d_out, int out_size, void* d_ws, size_t ws_size,
                              hipStream_t stream) {
  const float* x  = (const float*)d_in[0];
  const float* Wq = (const float*)d_in[1];
  const float* bq = (const float*)d_in[2];
  const float* Wk = (const float*)d_in[3];
  const float* bk = (const float*)d_in[4];
  const float* Wv = (const float*)d_in[5];
  const float* bv = (const float*)d_in[6];
  float* out = (float*)d_out;

  // Workspace layout (~134 MB total, all L2-resident):
  unsigned char* w = (unsigned char*)d_ws;
  unsigned short* xbf = (unsigned short*)w; w += (size_t)B_ * S_ * D_ * 2; // 16MB
  unsigned short* Wtq = (unsigned short*)w; w += (size_t)D_ * D_ * 2;      //  2MB
  unsigned short* Wtk = (unsigned short*)w; w += (size_t)D_ * D_ * 2;      //  2MB
  unsigned short* Wtv = (unsigned short*)w; w += (size_t)D_ * D_ * 2;      //  2MB
  unsigned short* Qbf = (unsigned short*)w; w += (size_t)B_ * S_ * D_ * 2; // 16MB
  unsigned short* Kbf = (unsigned short*)w; w += (size_t)B_ * S_ * D_ * 2; // 16MB
  unsigned short* Vt  = (unsigned short*)w; w += (size_t)B_ * D_ * S_ * 2; // 16MB
  float* scores = (float*)w;                 // B*S*S f32 = 64MB (reused as bf16 weights)

  // 1) bf16 conversions (W transposed for contiguous B-fragment loads).
  cvt_f32_to_bf16<<<4096, 256, 0, stream>>>(x, xbf, B_ * S_ * D_);
  transpose_cvt_w<<<(D_ * D_) / 256, 256, 0, stream>>>(Wq, Wtq);
  transpose_cvt_w<<<(D_ * D_) / 256, 256, 0, stream>>>(Wk, Wtk);
  transpose_cvt_w<<<(D_ * D_) / 256, 256, 0, stream>>>(Wv, Wtv);

  // 2) QKV projections. Q/K row-major bf16; V stored transposed [B][D][S].
  dim3 gp(S_ / 256, D_ / 64, B_);
  gemm_bf16_wmma<1><<<gp, 256, 0, stream>>>(
      xbf, D_, (size_t)S_ * D_, Wtq, D_, 0,
      Qbf, D_, (size_t)S_ * D_, bq, D_, 1.0f);
  gemm_bf16_wmma<1><<<gp, 256, 0, stream>>>(
      xbf, D_, (size_t)S_ * D_, Wtk, D_, 0,
      Kbf, D_, (size_t)S_ * D_, bk, D_, 1.0f);
  gemm_bf16_wmma<2><<<gp, 256, 0, stream>>>(
      xbf, D_, (size_t)S_ * D_, Wtv, D_, 0,
      Vt, S_, (size_t)D_ * S_, bv, D_, 1.0f);

  // 3) scores = Q @ K^T * 1/sqrt(D). K row-major == Bt layout directly.
  dim3 gs(S_ / 256, S_ / 64, B_);
  gemm_bf16_wmma<0><<<gs, 256, 0, stream>>>(
      Qbf, D_, (size_t)S_ * D_, Kbf, D_, (size_t)S_ * D_,
      scores, S_, (size_t)S_ * S_, nullptr, D_, 0.03125f);

  // 4) row softmax, bf16 weights written in place over the f32 rows.
  softmax_rows<<<B_ * S_, 256, 0, stream>>>(scores);

  // 5) out = weights @ V, using Vt so B-fragment loads stay contiguous.
  dim3 go(S_ / 256, D_ / 64, B_);
  gemm_bf16_wmma<0><<<go, 256, 0, stream>>>(
      (const unsigned short*)scores, 2 * S_, (size_t)2 * S_ * S_,
      Vt, S_, (size_t)D_ * S_,
      out, D_, (size_t)S_ * D_, nullptr, S_, 1.0f);
}
